// FasterRCNN_53541062312289
// MI455X (gfx1250) — matrix operand order repaired and checked by reading, not compiled, not run on hardware
//
#include <hip/hip_runtime.h>

// ---------------------------------------------------------------------------
// Faster R-CNN forward for MI455X (gfx1250, wave32, WMMA).
// Dense math via v_wmma_f32_16x16x32_f16 (f16 operands, f32 accumulate).
// f16 activations end-to-end; branch-free vectorized LDS staging; paired-k
// transpose staging (ds_store_b32); TDM tensor_load_to_lds path for the f16
// A operand (fc1), 6-arg builtin form (clang-23 toolchain).
// ---------------------------------------------------------------------------

typedef __attribute__((ext_vector_type(16))) _Float16 v16h;
typedef __attribute__((ext_vector_type(8)))  _Float16 v8h;
typedef __attribute__((ext_vector_type(2)))  _Float16 v2h;
typedef __attribute__((ext_vector_type(8)))  float    v8f;

#if __has_builtin(__builtin_amdgcn_tensor_load_to_lds) && __has_builtin(__builtin_amdgcn_s_wait_tensorcnt)
#define USE_TDM 1
typedef unsigned int v4u_t __attribute__((ext_vector_type(4)));
typedef int          v8i_t __attribute__((ext_vector_type(8)));
typedef int          v4i_t __attribute__((ext_vector_type(4)));
#else
#define USE_TDM 0
#endif

#define TILE_M 128
#define TILE_N 128
#define TILE_K 32
#define LDSH   40   // halves per LDS row: 32 data + 8 pad (80B, 16B aligned)

// ---- WMMA fragment loaders (per CDNA5 ISA 16-bit A 16x32 / B 32x16 layout) --
__device__ __forceinline__ v16h frag_a(const _Float16* Ash, int row, int half) {
  const v8h* p = (const v8h*)(Ash + row * LDSH);
  v8h lo = p[half];
  v8h hi = p[2 + half];
  return __builtin_shufflevector(lo, hi, 0,1,2,3,4,5,6,7,8,9,10,11,12,13,14,15);
}
__device__ __forceinline__ v16h frag_b(const _Float16* Bsh, int col, int half) {
  const v8h* p = (const v8h*)(Bsh + col * LDSH);
  v8h lo = p[half];
  v8h hi = p[2 + half];
  return __builtin_shufflevector(lo, hi, 0,1,2,3,4,5,6,7,8,9,10,11,12,13,14,15);
}

__device__ __forceinline__ v8h cvt8(float4 a, float4 b) {
  v8h h;
  h[0] = (_Float16)a.x; h[1] = (_Float16)a.y; h[2] = (_Float16)a.z; h[3] = (_Float16)a.w;
  h[4] = (_Float16)b.x; h[5] = (_Float16)b.y; h[6] = (_Float16)b.z; h[7] = (_Float16)b.w;
  return h;
}

#if USE_TDM
// Issue a TDM 2D tile load: 128 rows x 32 halves (row stride K halves) from
// `src` into padded LDS rows (16 DWORDs data + 4 DWORDs pad = LDSH halves).
__device__ __forceinline__ void tdm_load_a_tile(const _Float16* src, _Float16* lds, int K) {
  unsigned lds_addr = (unsigned)(size_t)lds;           // LDS aperture: low bits = LDS offset
  unsigned long long ga = (unsigned long long)(size_t)src;
  v4u_t g0;
  g0[0] = 1u;                                          // count=1, user descriptor
  g0[1] = lds_addr;                                    // lds_addr
  g0[2] = (unsigned)ga;                                // global_addr[31:0]
  g0[3] = (unsigned)((ga >> 32) & 0x01FFFFFFu) | (2u << 30);  // global_addr[56:32] | type=2
  v8i_t g1;
  g1[0] = (1 << 16)      // data_size = 1 -> 2 bytes
        | (1 << 20)      // pad_enable
        | (3 << 22)      // pad_interval: 16 DWORDs between pads
        | (3 << 25);     // pad_amount: 4 DWORDs
  g1[1] = (int)(((unsigned)K & 0xFFFFu) << 16);                    // tensor_dim0 low16
  g1[2] = (int)((((unsigned)K >> 16) & 0xFFFFu) | (128u << 16));   // tensor_dim0 hi | tensor_dim1 lo (=128)
  g1[3] = (int)(32u << 16);                                        // tensor_dim1 hi=0 | tile_dim0=32
  g1[4] = 128;                                                     // tile_dim1=128, tile_dim2=0
  g1[5] = K;                                                       // tensor_dim0_stride low32 (data units)
  g1[6] = 0;
  g1[7] = 0;
  v4i_t gz = {0, 0, 0, 0};
  v8i_t gz8 = {0, 0, 0, 0, 0, 0, 0, 0};
  __builtin_amdgcn_tensor_load_to_lds(g0, g1, gz, gz, gz8, 0);
}
#endif

// ---------------------------------------------------------------------------
// GEMM: C[M,N](f32) = A[M,K] * B[K,N]  (+bias_m / +bias_n, optional relu)
// ---------------------------------------------------------------------------
template <int AF16, int BF16>
__global__ __launch_bounds__(256) void gemm_wmma_kernel(
    const void* __restrict__ Av, const void* __restrict__ Bv,
    const float* __restrict__ bias_m, const float* __restrict__ bias_n,
    float* __restrict__ C, int M, int N, int K, int ldc, int relu)
{
  __shared__ alignas(16) _Float16 Ash[TILE_M * LDSH];
  __shared__ alignas(16) _Float16 Bsh[TILE_N * LDSH];

  const int tid  = threadIdx.x;
  const int lane = tid & 31;
  const int l16  = lane & 15;
  const int half = lane >> 4;
  const int wid  = tid >> 5;
  const int wm   = wid >> 1;
  const int wn   = wid & 1;
  const int tile_m = blockIdx.y * TILE_M;
  const int tile_n = blockIdx.x * TILE_N;

  const int ar  = tid >> 1;            // A fast: row 0..127
  const int ac0 = (tid & 1) * 16;      // A fast: 16-half chunk
  const int bk2 = (tid & 15) * 2;      // B fast: 2 consecutive k
  const int bn8 = (tid >> 4) * 8;      // B fast: 8-col group
  const int bkk = tid & 31;            // B slow: one k
  const int bn0 = (tid >> 5) * 16;     // B slow: 16-col group

  v8f acc[2][4];
  const v8f vzero = {0.f,0.f,0.f,0.f,0.f,0.f,0.f,0.f};
#pragma unroll
  for (int i = 0; i < 2; i++)
#pragma unroll
    for (int j = 0; j < 4; j++) acc[i][j] = vzero;

  for (int k0 = 0; k0 < K; k0 += TILE_K) {
    // ---------------- stage A (128 x 32, K-major) ----------------
    if (tile_m + TILE_M <= M && k0 + TILE_K <= K) {
      if (AF16) {
#if USE_TDM
        if (wid == 0) {
          tdm_load_a_tile((const _Float16*)Av + (size_t)tile_m * K + k0, Ash, K);
          __builtin_amdgcn_s_wait_tensorcnt(0);
        }
#else
        const v8h* g = (const v8h*)((const _Float16*)Av + (size_t)(tile_m + ar) * K + k0 + ac0);
        *(v8h*)(Ash + ar * LDSH + ac0)     = g[0];
        *(v8h*)(Ash + ar * LDSH + ac0 + 8) = g[1];
        if (k0 + TILE_K < K) __builtin_prefetch((const _Float16*)Av + (size_t)(tile_m + ar) * K + k0 + TILE_K + ac0, 0, 0);
#endif
      } else {
        const float4* g = (const float4*)((const float*)Av + (size_t)(tile_m + ar) * K + k0 + ac0);
        float4 f0 = g[0], f1 = g[1], f2 = g[2], f3 = g[3];
        *(v8h*)(Ash + ar * LDSH + ac0)     = cvt8(f0, f1);
        *(v8h*)(Ash + ar * LDSH + ac0 + 8) = cvt8(f2, f3);
        if (k0 + TILE_K < K) __builtin_prefetch((const float*)Av + (size_t)(tile_m + ar) * K + k0 + TILE_K + ac0, 0, 0);
      }
    } else {
#pragma unroll
      for (int i = 0; i < 16; i++) {
        int idx = i * 256 + tid;
        int r = idx >> 5, c = idx & 31;
        int gr = tile_m + r, gc = k0 + c;
        int ok = (gr < M) & (gc < K);
        size_t off = (size_t)min(gr, M - 1) * K + min(gc, K - 1);
        _Float16 v = AF16 ? ((const _Float16*)Av)[off] : (_Float16)((const float*)Av)[off];
        Ash[r * LDSH + c] = ok ? v : (_Float16)0.f;
      }
    }
    // ------- stage B (32 x 128 -> N-major LDS), paired-k b32 stores -------
    if (tile_n + TILE_N <= N && k0 + TILE_K <= K) {
      v8h r0, r1;
      if (BF16) {
        r0 = *(const v8h*)((const _Float16*)Bv + (size_t)(k0 + bk2)     * N + tile_n + bn8);
        r1 = *(const v8h*)((const _Float16*)Bv + (size_t)(k0 + bk2 + 1) * N + tile_n + bn8);
        if (k0 + TILE_K < K) __builtin_prefetch((const _Float16*)Bv + (size_t)(k0 + TILE_K + bk2) * N + tile_n + bn8, 0, 0);
      } else {
        const float4* ga = (const float4*)((const float*)Bv + (size_t)(k0 + bk2)     * N + tile_n + bn8);
        const float4* gb = (const float4*)((const float*)Bv + (size_t)(k0 + bk2 + 1) * N + tile_n + bn8);
        float4 a0 = ga[0], a1 = ga[1], b0 = gb[0], b1 = gb[1];
        r0 = cvt8(a0, a1);
        r1 = cvt8(b0, b1);
        if (k0 + TILE_K < K) __builtin_prefetch((const float*)Bv + (size_t)(k0 + TILE_K + bk2) * N + tile_n + bn8, 0, 0);
      }
#pragma unroll
      for (int j = 0; j < 8; j++) {
        v2h pk; pk[0] = r0[j]; pk[1] = r1[j];
        *(v2h*)(Bsh + (bn8 + j) * LDSH + bk2) = pk;
      }
    } else {
      int gk = k0 + bkk;
      int kok = (gk < K);
      int gkc = min(gk, K - 1);
#pragma unroll
      for (int i = 0; i < 16; i++) {
        int gn = tile_n + bn0 + i;
        int ok = kok & (gn < N);
        size_t off = (size_t)gkc * N + min(gn, N - 1);
        _Float16 v = BF16 ? ((const _Float16*)Bv)[off] : (_Float16)((const float*)Bv)[off];
        Bsh[(bn0 + i) * LDSH + bkk] = ok ? v : (_Float16)0.f;
      }
    }
    __syncthreads();

    v16h bf[4];
#pragma unroll
    for (int ni = 0; ni < 4; ni++)
      bf[ni] = frag_b(Bsh, wn * 64 + ni * 16 + l16, half);
#pragma unroll
    for (int mi = 0; mi < 2; mi++) {
      v16h af = frag_a(Ash, wm * 32 + mi * 16 + l16, half);
#pragma unroll
      for (int ni = 0; ni < 4; ni++)
        acc[mi][ni] = __builtin_amdgcn_wmma_f32_16x16x32_f16(
            false, af, false, bf[ni], (short)0, acc[mi][ni], false, false);
    }
    __syncthreads();
  }

#pragma unroll
  for (int mi = 0; mi < 2; mi++)
#pragma unroll
    for (int ni = 0; ni < 4; ni++) {
      int col = tile_n + wn * 64 + ni * 16 + l16;
      if (col >= N) continue;
      float bn = bias_n ? bias_n[col] : 0.f;
      int rbase = tile_m + wm * 32 + mi * 16 + half * 8;
#pragma unroll
      for (int v = 0; v < 8; v++) {
        int r = rbase + v;
        if (r < M) {
          float val = acc[mi][ni][v] + bn + (bias_m ? bias_m[r] : 0.f);
          if (relu) val = fmaxf(val, 0.f);
          C[(size_t)r * ldc + col] = val;
        }
      }
    }
}

// ---------------------------------------------------------------------------
// Implicit-GEMM conv: out(f16)[Cout, OH*OW] = W(f32)[Cout, Cin*KH*KW] x im2col(in f16)
// OH*OW is always a multiple of 128 here, so no N-guard needed.
// ---------------------------------------------------------------------------
__global__ __launch_bounds__(256) void conv_wmma_kernel(
    const _Float16* __restrict__ in,
    const float* __restrict__ W,
    const float* __restrict__ bias,
    _Float16* __restrict__ out,
    int Cin, int IH, int IW,
    int Cout, int OH, int OW,
    int KH, int KW, int stride, int pad, int relu)
{
  __shared__ alignas(16) _Float16 Ash[TILE_M * LDSH];
  __shared__ alignas(16) _Float16 Bsh[TILE_N * LDSH];

  const int tid  = threadIdx.x;
  const int lane = tid & 31;
  const int l16  = lane & 15;
  const int half = lane >> 4;
  const int wid  = tid >> 5;
  const int wm   = wid >> 1;
  const int wn   = wid & 1;
  const int tile_m = blockIdx.y * TILE_M;
  const int tile_n = blockIdx.x * TILE_N;

  const int KHKW = KH * KW;
  const int K    = Cin * KHKW;
  const int Npix = OH * OW;

  const int ar  = tid >> 1;
  const int ac0 = (tid & 1) * 16;
  const int bk2 = (tid & 15) * 2;      // 2 consecutive k
  const int bn8 = (tid >> 4) * 8;      // 8 consecutive pixels

  v8f acc[2][4];
  const v8f vzero = {0.f,0.f,0.f,0.f,0.f,0.f,0.f,0.f};
#pragma unroll
  for (int i = 0; i < 2; i++)
#pragma unroll
    for (int j = 0; j < 4; j++) acc[i][j] = vzero;

  for (int k0 = 0; k0 < K; k0 += TILE_K) {
    // ---- stage weights ----
    if (tile_m + TILE_M <= Cout && k0 + TILE_K <= K) {
      const float4* g = (const float4*)(W + (size_t)(tile_m + ar) * K + k0 + ac0);
      float4 f0 = g[0], f1 = g[1], f2 = g[2], f3 = g[3];
      *(v8h*)(Ash + ar * LDSH + ac0)     = cvt8(f0, f1);
      *(v8h*)(Ash + ar * LDSH + ac0 + 8) = cvt8(f2, f3);
      if (k0 + TILE_K < K) __builtin_prefetch(W + (size_t)(tile_m + ar) * K + k0 + TILE_K + ac0, 0, 0);
    } else {
#pragma unroll
      for (int i = 0; i < 16; i++) {
        int idx = i * 256 + tid;
        int r = idx >> 5, c = idx & 31;
        int gr = tile_m + r, gc = k0 + c;
        int ok = (gr < Cout) & (gc < K);
        size_t off = (size_t)min(gr, Cout - 1) * K + min(gc, K - 1);
        float v = W[off];
        Ash[r * LDSH + c] = ok ? (_Float16)v : (_Float16)0.f;
      }
    }
    // ---- stage im2col: thread owns 2 consecutive k x 8 consecutive pixels ----
    {
      int gka = k0 + bk2, gkb = gka + 1;
      int oka = (gka < K), okb = (gkb < K);
      int gkac = min(gka, K - 1), gkbc = min(gkb, K - 1);
      int cia = gkac / KHKW, rema = gkac - cia * KHKW;
      int fya = rema / KW,   fxa  = rema - fya * KW;
      int cib = gkbc / KHKW, remb = gkbc - cib * KHKW;
      int fyb = remb / KW,   fxb  = remb - fyb * KW;
#pragma unroll
      for (int i = 0; i < 8; i++) {
        int gn = tile_n + bn8 + i;        // < Npix by construction
        int oy = gn / OW, ox = gn - oy * OW;
        int sy = oy * stride - pad, sx = ox * stride - pad;
        int iya = sy + fya, ixa = sx + fxa;
        int iyb = sy + fyb, ixb = sx + fxb;
        int oa = oka & (iya >= 0) & (iya < IH) & (ixa >= 0) & (ixa < IW);
        int ob = okb & (iyb >= 0) & (iyb < IH) & (ixb >= 0) & (ixb < IW);
        _Float16 va = in[((size_t)cia * IH + min(max(iya, 0), IH - 1)) * IW + min(max(ixa, 0), IW - 1)];
        _Float16 vb = in[((size_t)cib * IH + min(max(iyb, 0), IH - 1)) * IW + min(max(ixb, 0), IW - 1)];
        v2h pk; pk[0] = oa ? va : (_Float16)0.f; pk[1] = ob ? vb : (_Float16)0.f;
        *(v2h*)(Bsh + (bn8 + i) * LDSH + bk2) = pk;
      }
    }
    __syncthreads();

    v16h bf[4];
#pragma unroll
    for (int ni = 0; ni < 4; ni++)
      bf[ni] = frag_b(Bsh, wn * 64 + ni * 16 + l16, half);
#pragma unroll
    for (int mi = 0; mi < 2; mi++) {
      v16h af = frag_a(Ash, wm * 32 + mi * 16 + l16, half);
#pragma unroll
      for (int ni = 0; ni < 4; ni++)
        acc[mi][ni] = __builtin_amdgcn_wmma_f32_16x16x32_f16(
            false, af, false, bf[ni], (short)0, acc[mi][ni], false, false);
    }
    __syncthreads();
  }

#pragma unroll
  for (int mi = 0; mi < 2; mi++)
#pragma unroll
    for (int ni = 0; ni < 4; ni++) {
      int col = tile_n + wn * 64 + ni * 16 + l16;
      int rbase = tile_m + wm * 32 + mi * 16 + half * 8;
#pragma unroll
      for (int v = 0; v < 8; v++) {
        int r = rbase + v;
        if (r < Cout) {
          float val = acc[mi][ni][v] + (bias ? bias[r] : 0.f);
          if (relu) val = fmaxf(val, 0.f);
          out[(size_t)r * Npix + col] = (_Float16)val;
        }
      }
    }
}

// ---------------------------------------------------------------------------
// elementwise f32 -> f16
// ---------------------------------------------------------------------------
__global__ void f32_to_f16_kernel(const float* __restrict__ in, _Float16* __restrict__ out, int n)
{
  int i = blockIdx.x * 256 + threadIdx.x;
  if (i < n) out[i] = (_Float16)in[i];
}

// ---------------------------------------------------------------------------
// maxpool 3x3 s2 p1 : [64,256,256] -> [64,128,128] (f16)
// ---------------------------------------------------------------------------
__global__ void maxpool_kernel(const _Float16* __restrict__ in, _Float16* __restrict__ out)
{
  int i = blockIdx.x * 256 + threadIdx.x;
  if (i >= 64 * 128 * 128) return;
  int c = i >> 14, rem = i & 16383;
  int oy = rem >> 7, ox = rem & 127;
  float m = -INFINITY;
#pragma unroll
  for (int fy = 0; fy < 3; fy++) {
    int iy = oy * 2 - 1 + fy;
    if (iy < 0 || iy >= 256) continue;
#pragma unroll
    for (int fx = 0; fx < 3; fx++) {
      int ix = ox * 2 - 1 + fx;
      if (ix < 0 || ix >= 256) continue;
      m = fmaxf(m, (float)in[(c << 16) + (iy << 8) + ix]);
    }
  }
  out[i] = (_Float16)m;
}

// ---------------------------------------------------------------------------
// anchor decode (torchvision flattening: (y,x,a), anchor innermost)
// ---------------------------------------------------------------------------
__global__ void decode_kernel(const float* __restrict__ obj,
                              const float* __restrict__ reg,
                              float* __restrict__ boxes,
                              float* __restrict__ scores)
{
  int i = blockIdx.x * 256 + threadIdx.x;
  if (i >= 12288) return;
  int p = i / 12, a = i - p * 12;
  int y = p >> 5, x = p & 31;
  int r = a >> 2, s = a & 3;
  const float hr_t[3] = {0.70710678118654752f, 1.0f, 1.41421356237309515f};
  const float sc_t[4] = {32.f, 64.f, 128.f, 256.f};
  float hr = hr_t[r], sc = sc_t[s];
  float hs = hr * sc, ws = sc / hr;
  float cxa = x * 16.f, cya = y * 16.f;
  float d0 = reg[(a * 4 + 0) * 1024 + p];
  float d1 = reg[(a * 4 + 1) * 1024 + p];
  float d2 = reg[(a * 4 + 2) * 1024 + p];
  float d3 = reg[(a * 4 + 3) * 1024 + p];
  const float CLAMP = 4.135166556742356f;  // log(1000/16)
  float dw = fminf(d2, CLAMP), dh = fminf(d3, CLAMP);
  float cx = d0 * ws + cxa, cy = d1 * hs + cya;
  float w = ws * expf(dw), h = hs * expf(dh);
  boxes[i * 4 + 0] = fminf(fmaxf(cx - 0.5f * w, 0.f), 512.f);
  boxes[i * 4 + 1] = fminf(fmaxf(cy - 0.5f * h, 0.f), 512.f);
  boxes[i * 4 + 2] = fminf(fmaxf(cx + 0.5f * w, 0.f), 512.f);
  boxes[i * 4 + 3] = fminf(fmaxf(cy + 0.5f * h, 0.f), 512.f);
  scores[i] = obj[a * 1024 + p];
}

// ---------------------------------------------------------------------------
// top-1000 via iterative block-wide argmax (deterministic lower-index ties)
// ---------------------------------------------------------------------------
__global__ __launch_bounds__(1024) void topk_kernel(const float* __restrict__ scores,
                                                    float* __restrict__ tmp,
                                                    int* __restrict__ topk)
{
  __shared__ float sval[1024];
  __shared__ int   sidx[1024];
  int tid = threadIdx.x;
  for (int j = tid; j < 12288; j += 1024) tmp[j] = scores[j];
  __syncthreads();
  for (int t = 0; t < 1000; t++) {
    float best = -INFINITY;
    int   bi   = 0x7fffffff;
    for (int j = tid; j < 12288; j += 1024) {
      float v = tmp[j];
      if (v > best) { best = v; bi = j; }
    }
    sval[tid] = best; sidx[tid] = bi;
    __syncthreads();
    for (int s = 512; s > 0; s >>= 1) {
      if (tid < s) {
        float ov = sval[tid + s]; int oi = sidx[tid + s];
        if (ov > sval[tid] || (ov == sval[tid] && oi < sidx[tid])) {
          sval[tid] = ov; sidx[tid] = oi;
        }
      }
      __syncthreads();
    }
    if (tid == 0) { topk[t] = sidx[0]; tmp[sidx[0]] = -INFINITY; }
    __syncthreads();
  }
}

// ---------------------------------------------------------------------------
// greedy NMS over 1000 sorted candidates -> 256 proposals
// ---------------------------------------------------------------------------
__global__ __launch_bounds__(1024) void nms_kernel(const float* __restrict__ boxes,
                                                   const int* __restrict__ topk,
                                                   float* __restrict__ props)
{
  __shared__ float bx[1000][4];
  __shared__ int   keep[1000];
  int tid = threadIdx.x;
  if (tid < 1000) {
    int id = topk[tid];
    bx[tid][0] = boxes[id * 4 + 0];
    bx[tid][1] = boxes[id * 4 + 1];
    bx[tid][2] = boxes[id * 4 + 2];
    bx[tid][3] = boxes[id * 4 + 3];
    keep[tid] = 1;
  }
  __syncthreads();
  for (int i = 0; i < 1000; i++) {
    if (tid < 1000 && tid > i && keep[i] && keep[tid]) {
      float aa = (bx[i][2] - bx[i][0]) * (bx[i][3] - bx[i][1]);
      float ab = (bx[tid][2] - bx[tid][0]) * (bx[tid][3] - bx[tid][1]);
      float iw = fmaxf(fminf(bx[i][2], bx[tid][2]) - fmaxf(bx[i][0], bx[tid][0]), 0.f);
      float ih = fmaxf(fminf(bx[i][3], bx[tid][3]) - fmaxf(bx[i][1], bx[tid][1]), 0.f);
      float inter = iw * ih;
      float iou = inter / (aa + ab - inter + 1e-9f);
      if (iou > 0.7f) keep[tid] = 0;
    }
    __syncthreads();
  }
  if (tid == 0) {
    int cnt = 0;
    for (int j = 0; j < 1000 && cnt < 256; j++)
      if (keep[j]) {
        props[cnt * 4 + 0] = bx[j][0]; props[cnt * 4 + 1] = bx[j][1];
        props[cnt * 4 + 2] = bx[j][2]; props[cnt * 4 + 3] = bx[j][3];
        cnt++;
      }
    for (int j = 0; j < 1000 && cnt < 256; j++)
      if (!keep[j]) {
        props[cnt * 4 + 0] = bx[j][0]; props[cnt * 4 + 1] = bx[j][1];
        props[cnt * 4 + 2] = bx[j][2]; props[cnt * 4 + 3] = bx[j][3];
        cnt++;
      }
  }
}

// ---------------------------------------------------------------------------
// ROI-align (one bilinear sample per 7x7 cell), f16 feat -> f16 pooled
// pooled layout [R, 2048*49], index c*49 + py*7 + px
// ---------------------------------------------------------------------------
__global__ void roi_align_kernel(const _Float16* __restrict__ feat,
                                 const float* __restrict__ props,
                                 _Float16* __restrict__ pooled)
{
  int blk = blockIdx.x;
  int rix = blk / 49, cell = blk - rix * 49;
  int py = cell / 7, px = cell - py * 7;
  const float inv16 = 1.f / 16.f;
  float b0 = props[rix * 4 + 0] * inv16;
  float b1 = props[rix * 4 + 1] * inv16;
  float b2 = props[rix * 4 + 2] * inv16;
  float b3 = props[rix * 4 + 3] * inv16;
  float gy = (py + 0.5f) / 7.f, gx = (px + 0.5f) / 7.f;
  float yy = b1 + gy * (b3 - b1) - 0.5f;
  float xx = b0 + gx * (b2 - b0) - 0.5f;
  float y0f = floorf(yy), x0f = floorf(xx);
  float wy = yy - y0f, wx = xx - x0f;
  int y0 = min(max((int)y0f, 0), 31);
  int y1 = min(y0 + 1, 31);
  int x0 = min(max((int)x0f, 0), 31);
  int x1 = min(x0 + 1, 31);
  float w00 = (1.f - wy) * (1.f - wx), w01 = (1.f - wy) * wx;
  float w10 = wy * (1.f - wx),         w11 = wy * wx;
  for (int c = threadIdx.x; c < 2048; c += 256) {
    const _Float16* f = feat + (size_t)c * 1024;
    float v = (float)f[y0 * 32 + x0] * w00 + (float)f[y0 * 32 + x1] * w01 +
              (float)f[y1 * 32 + x0] * w10 + (float)f[y1 * 32 + x1] * w11;
    pooled[(size_t)rix * 100352 + c * 49 + cell] = (_Float16)v;
  }
}

// ---------------------------------------------------------------------------
// host-side launcher
// ---------------------------------------------------------------------------
static inline int cdiv_i(int a, int b) { return (a + b - 1) / b; }

extern "C" void kernel_launch(void* const* d_in, const int* in_sizes, int n_in,
                              void* d_out, int out_size, void* d_ws, size_t ws_size,
                              hipStream_t stream)
{
  (void)in_sizes; (void)n_in; (void)out_size; (void)ws_size;

  const float* images = (const float*)d_in[0];
  const float* w_stem = (const float*)d_in[1];
  const float* w1     = (const float*)d_in[2];
  const float* w2     = (const float*)d_in[3];
  const float* w3     = (const float*)d_in[4];
  const float* w4     = (const float*)d_in[5];
  const float* w_rpn  = (const float*)d_in[6];
  const float* b_rpn  = (const float*)d_in[7];
  const float* w_obj  = (const float*)d_in[8];
  const float* b_obj  = (const float*)d_in[9];
  const float* w_reg  = (const float*)d_in[10];
  const float* b_regc = (const float*)d_in[11];
  const float* fc1_w  = (const float*)d_in[12];
  const float* fc1_b  = (const float*)d_in[13];
  const float* fc2_w  = (const float*)d_in[14];
  const float* fc2_b  = (const float*)d_in[15];
  const float* cls_w  = (const float*)d_in[16];
  const float* cls_b  = (const float*)d_in[17];
  const float* box_w  = (const float*)d_in[18];
  const float* box_b  = (const float*)d_in[19];
  float* out = (float*)d_out;

  // workspace layout (byte offsets); f16 activations; peak ~70 MB
  char* wsb = (char*)d_ws;
  _Float16* S      = (_Float16*)(wsb + 0);          // 4,194,304 h : stem out / x1 / feat
  _Float16* P      = (_Float16*)(wsb + 8388608);    // 1,048,576 h : pool out / x3
  _Float16* B2     = (_Float16*)(wsb + 10485760);   // 2,097,152 h : x2 / rpn-t
  _Float16* IMG    = (_Float16*)(wsb + 14680064);   //   786,432 h : images f16
  float* obj       = (float*)(wsb + 16252928);      // 12*1024
  float* reg       = (float*)(wsb + 16302080);      // 48*1024
  float* boxes     = (float*)(wsb + 16498688);      // 12288*4
  float* scores    = (float*)(wsb + 16695296);      // 12288
  float* sctmp     = (float*)(wsb + 16744448);      // 12288
  int*   topkb     = (int*)(wsb + 16793600);        // 1024
  float* props     = (float*)(wsb + 16797696);      // 1024
  float* h1        = (float*)(wsb + 16801792);      // 256*1024
  float* h2        = (float*)(wsb + 17850368);      // 256*1024
  _Float16* pooled = (_Float16*)(wsb + 18898944);   // 256*100352 h

  // --- images -> f16 ---
  f32_to_f16_kernel<<<dim3(cdiv_i(786432, 256)), 256, 0, stream>>>(images, IMG, 786432);

  // --- backbone ---
  conv_wmma_kernel<<<dim3(cdiv_i(65536,128), 1), 256, 0, stream>>>(
      IMG, w_stem, nullptr, S, 3, 512, 512, 64, 256, 256, 7, 7, 2, 3, 1);
  maxpool_kernel<<<dim3(cdiv_i(64*128*128,256)), 256, 0, stream>>>(S, P);
  conv_wmma_kernel<<<dim3(cdiv_i(16384,128), cdiv_i(256,128)), 256, 0, stream>>>(
      P, w1, nullptr, S, 64, 128, 128, 256, 128, 128, 3, 3, 1, 1, 1);
  conv_wmma_kernel<<<dim3(cdiv_i(4096,128), cdiv_i(512,128)), 256, 0, stream>>>(
      S, w2, nullptr, B2, 256, 128, 128, 512, 64, 64, 3, 3, 2, 1, 1);
  conv_wmma_kernel<<<dim3(cdiv_i(1024,128), cdiv_i(1024,128)), 256, 0, stream>>>(
      B2, w3, nullptr, P, 512, 64, 64, 1024, 32, 32, 3, 3, 2, 1, 1);
  conv_wmma_kernel<<<dim3(cdiv_i(1024,128), cdiv_i(2048,128)), 256, 0, stream>>>(
      P, w4, nullptr, S, 1024, 32, 32, 2048, 32, 32, 3, 3, 1, 1, 1);   // feat -> S

  // --- RPN head ---
  conv_wmma_kernel<<<dim3(cdiv_i(1024,128), cdiv_i(2048,128)), 256, 0, stream>>>(
      S, w_rpn, b_rpn, B2, 2048, 32, 32, 2048, 32, 32, 3, 3, 1, 1, 1); // t -> B2
  gemm_wmma_kernel<0,1><<<dim3(cdiv_i(1024,128), 1), 256, 0, stream>>>(
      w_obj, B2, b_obj, nullptr, obj, 12, 1024, 2048, 1024, 0);
  gemm_wmma_kernel<0,1><<<dim3(cdiv_i(1024,128), 1), 256, 0, stream>>>(
      w_reg, B2, b_regc, nullptr, reg, 48, 1024, 2048, 1024, 0);

  // --- proposals ---
  decode_kernel<<<dim3(cdiv_i(12288,256)), 256, 0, stream>>>(obj, reg, boxes, scores);
  topk_kernel<<<1, 1024, 0, stream>>>(scores, sctmp, topkb);
  nms_kernel<<<1, 1024, 0, stream>>>(boxes, topkb, props);

  // --- ROI heads ---
  roi_align_kernel<<<dim3(256 * 49), 256, 0, stream>>>(S, props, pooled);
  gemm_wmma_kernel<1,0><<<dim3(cdiv_i(1024,128), cdiv_i(256,128)), 256, 0, stream>>>(
      pooled, fc1_w, nullptr, fc1_b, h1, 256, 1024, 100352, 1024, 1);
  gemm_wmma_kernel<0,0><<<dim3(cdiv_i(1024,128), cdiv_i(256,128)), 256, 0, stream>>>(
      h1, fc2_w, nullptr, fc2_b, h2, 256, 1024, 1024, 1024, 1);
  gemm_wmma_kernel<0,0><<<dim3(cdiv_i(91,128), cdiv_i(256,128)), 256, 0, stream>>>(
      h2, cls_w, nullptr, cls_b, out, 256, 91, 1024, 455, 0);
  gemm_wmma_kernel<0,0><<<dim3(cdiv_i(364,128), cdiv_i(256,128)), 256, 0, stream>>>(
      h2, box_w, nullptr, box_b, out + 91, 256, 364, 1024, 455, 0);
}
